// EncoderSRNN_6253472383206
// MI455X (gfx1250) — matrix operand order repaired and checked by reading, not compiled, not run on hardware
//
#include <hip/hip_runtime.h>
#include <hip/hip_bf16.h>

// ---------------------------------------------------------------------------
// EncoderSRNN (stack-augmented GRU) for gfx1250 / MI455X
//   T=512 B=64 H=512 V=10000 N=2 S=32 E=64 NACT=3
// Strategy:
//   1) prep:     bf16-transpose W_ih/W_hh into [K][N] layout; copy stacks.
//   2) gi_gemm:  GI[T*B,3H] = emb[inputs] @ W_ih^T + b_ih  (bf16 WMMA, f32 acc)
//   3) per step: step_stack (VALU) + step_h (bf16 WMMA recurrent GEMM + gates)
//   4) finalize: copy final stacks into d_out.
// ---------------------------------------------------------------------------

typedef __attribute__((ext_vector_type(16))) __bf16 v16bf;
typedef __attribute__((ext_vector_type(8)))  float  v8f;

#define CT 512
#define CB 64
#define CH 512
#define CN 2
#define CS 32
#define CE 64
#define C3H 1536

// ---- WMMA fragment helpers (CDNA5 16x16x32 bf16 layouts) -------------------

// A (MxK=16x32, 16-bit): lane<16 -> M=lane,   elems 0..7 = K 0..7,  8..15 = K 16..23
//                        lane>=16 -> M=lane-16, elems 0..7 = K 8..15, 8..15 = K 24..31
__device__ __forceinline__ v16bf load_a_frag_f32(const float* rowp, int kk, int lane) {
    const float* p = rowp + kk + ((lane & 16) ? 8 : 0);
    v16bf a;
#pragma unroll
    for (int i = 0; i < 8; ++i) {
        a[i]     = (__bf16)p[i];
        a[i + 8] = (__bf16)p[i + 16];
    }
    return a;
}

// B (KxN=32x16, 16-bit): lane<16 -> N=lane, elems 0..7 = K 0..7, 8..15 = K 16..23;
//                        lane>=16 -> N=lane-16, elems 0..7 = K 8..15, 8..15 = K 24..31
// Bt stored K-major: Bt[k*ldn + n].
__device__ __forceinline__ v16bf load_b_frag(const __bf16* Bt, int ldn, int krow, int n0) {
    const int lane = krow & 31;
    const int kk = krow & ~31;
    const __bf16* p = Bt + (size_t)(kk + ((lane & 16) ? 8 : 0)) * ldn + n0 + (lane & 15);
    v16bf b;
#pragma unroll
    for (int i = 0; i < 8; ++i) {
        b[i]     = p[(size_t)i * ldn];
        b[i + 8] = p[(size_t)(i + 16) * ldn];
    }
    return b;
}

__device__ __forceinline__ float sigmoidf_(float x) {
    return 1.0f / (1.0f + __expf(-x));
}

// ---- prep: bf16 transposed weights + stack init ----------------------------
__global__ void srnn_prep_kernel(const float* __restrict__ W_ih,
                                 const float* __restrict__ W_hh,
                                 const float* __restrict__ stacks_in,
                                 __bf16* __restrict__ Wih_t,
                                 __bf16* __restrict__ Whh_t,
                                 float* __restrict__ st0) {
    const int WT = CH * C3H;                 // 786432
    const int ST = CB * CN * CS * CE;        // 262144
    const int total = 2 * WT + ST;
    for (int i = blockIdx.x * blockDim.x + threadIdx.x; i < total;
         i += gridDim.x * blockDim.x) {
        if (i < WT) {
            int k = i / C3H, n = i % C3H;
            Wih_t[i] = (__bf16)W_ih[(size_t)n * CH + k];
        } else if (i < 2 * WT) {
            int j = i - WT;
            int k = j / C3H, n = j % C3H;
            Whh_t[j] = (__bf16)W_hh[(size_t)n * CH + k];
        } else {
            int j = i - 2 * WT;
            st0[j] = stacks_in[j];
        }
    }
}

// ---- gi_gemm: GI[T*B, 3H] = gather(emb, inputs) @ Wih_t + b_ih -------------
// One wave computes a 16x64 output strip (4 N-tiles sharing the A fragment).
__global__ void srnn_gi_gemm_kernel(const int* __restrict__ inputs,
                                    const float* __restrict__ emb,
                                    const __bf16* __restrict__ Wih_t,
                                    const float* __restrict__ b_ih,
                                    __bf16* __restrict__ GI) {
    const int lane = threadIdx.x & 31;
    const int warp = threadIdx.x >> 5;
    const int wid  = blockIdx.x * 8 + warp;          // 49152 wave jobs
    const int mg = wid / 24;                         // 2048 M tiles
    const int ng = wid % 24;                         // 24 N groups of 64
    const int m0 = mg * 16;
    const int n0 = ng * 64;

    const int mrow = m0 + (lane & 15);
    const float* rowp = emb + (size_t)inputs[mrow] * CH;

    v8f acc[4];
#pragma unroll
    for (int j = 0; j < 4; ++j) {
        float bv = b_ih[n0 + 16 * j + (lane & 15)];
#pragma unroll
        for (int r = 0; r < 8; ++r) acc[j][r] = bv;
    }

    for (int kk = 0; kk < CH; kk += 32) {
        v16bf a = load_a_frag_f32(rowp, kk, lane);
#pragma unroll
        for (int j = 0; j < 4; ++j) {
            v16bf b = load_b_frag(Wih_t, C3H, kk + lane, n0 + 16 * j);
            acc[j] = __builtin_amdgcn_wmma_f32_16x16x32_bf16(
                false, a, false, b, (short)0, acc[j], false, false);
        }
    }

    const int rbase = m0 + ((lane & 16) ? 8 : 0);
    const int cb = lane & 15;
#pragma unroll
    for (int j = 0; j < 4; ++j)
#pragma unroll
        for (int r = 0; r < 8; ++r)
            GI[(size_t)(rbase + r) * C3H + n0 + 16 * j + cb] = (__bf16)acc[j][r];
}

// ---- step_h: new_h = GRU(h_prev, gi_t) ------------------------------------
// 128 waves: each computes one 16x16 tile of new_h; needs the r/z/n column
// tiles of gh (3 accumulators) plus the matching GI columns.
__global__ void srnn_step_h_kernel(const float* __restrict__ h_prev,
                                   const __bf16* __restrict__ GI_t,
                                   const __bf16* __restrict__ Whh_t,
                                   const float* __restrict__ b_hh,
                                   float* __restrict__ out_h,
                                   float* __restrict__ out_hfinal) {
    const int lane = threadIdx.x & 31;
    const int warp = threadIdx.x >> 5;
    const int wid  = blockIdx.x * 8 + warp;   // 0..127
    const int mt = wid & 3;                   // 4 M tiles (B=64)
    const int nt = wid >> 2;                  // 32 N tiles (H=512)
    const int m0 = mt * 16;
    const int n0 = nt * 16;

    const float* rowp = h_prev + (size_t)(m0 + (lane & 15)) * CH;

    v8f acc[3];
#pragma unroll
    for (int g = 0; g < 3; ++g) {
        float bv = b_hh[g * CH + n0 + (lane & 15)];
#pragma unroll
        for (int r = 0; r < 8; ++r) acc[g][r] = bv;
    }

    for (int kk = 0; kk < CH; kk += 32) {
        v16bf a = load_a_frag_f32(rowp, kk, lane);
#pragma unroll
        for (int g = 0; g < 3; ++g) {
            v16bf b = load_b_frag(Whh_t, C3H, kk + lane, g * CH + n0);
            acc[g] = __builtin_amdgcn_wmma_f32_16x16x32_bf16(
                false, a, false, b, (short)0, acc[g], false, false);
        }
    }

    const int rbase = m0 + ((lane & 16) ? 8 : 0);
    const int col = n0 + (lane & 15);
#pragma unroll
    for (int r = 0; r < 8; ++r) {
        const int row = rbase + r;
        const size_t gb = (size_t)row * C3H + col;
        float gir = (float)GI_t[gb];
        float giz = (float)GI_t[gb + CH];
        float gin = (float)GI_t[gb + 2 * CH];
        float rg = sigmoidf_(gir + acc[0][r]);
        float zg = sigmoidf_(giz + acc[1][r]);
        float ng = tanhf(gin + rg * acc[2][r]);
        float hv = h_prev[(size_t)row * CH + col];
        float nh = (1.0f - zg) * ng + zg * hv;
        out_h[(size_t)row * CH + col] = nh;
        if (out_hfinal) out_hfinal[(size_t)row * CH + col] = nh;
    }
}

// ---- step_stack: act softmax, push_vals, stack shift/blend -----------------
// One 64-thread block per (b, n) pair. Small GEMVs -> plain VALU from LDS.
__global__ void srnn_step_stack_kernel(const float* __restrict__ h_prev,
                                       const float* __restrict__ W_act,
                                       const float* __restrict__ b_act,
                                       const float* __restrict__ W_stk,
                                       const float* __restrict__ b_stk,
                                       const float* __restrict__ empty_e,
                                       const float* __restrict__ st_in,
                                       float* __restrict__ st_out) {
    const int bIdx = blockIdx.x >> 1;   // batch
    const int nIdx = blockIdx.x & 1;    // stack id
    const int tid  = threadIdx.x;       // 0..63

    __shared__ float sh[CH];
    __shared__ float slog[3];
    __shared__ float sprob[3];
    __shared__ float spush[CE];

    for (int i = tid; i < CH; i += 64) sh[i] = h_prev[(size_t)bIdx * CH + i];
    __syncthreads();

    // push value row tid
    {
        const float* wr = W_stk + (size_t)(nIdx * CE + tid) * CH;
        float a = b_stk[nIdx * CE + tid];
        for (int k = 0; k < CH; ++k) a += sh[k] * wr[k];
        spush[tid] = tanhf(a);
    }
    // action logits (threads 0..2)
    if (tid < 3) {
        const float* wa = W_act + (size_t)(nIdx * 3 + tid) * CH;
        float a = b_act[nIdx * 3 + tid];
        for (int k = 0; k < CH; ++k) a += sh[k] * wa[k];
        slog[tid] = a;
    }
    __syncthreads();
    if (tid == 0) {
        float m = fmaxf(slog[0], fmaxf(slog[1], slog[2]));
        float e0 = __expf(slog[0] - m);
        float e1 = __expf(slog[1] - m);
        float e2 = __expf(slog[2] - m);
        float inv = 1.0f / (e0 + e1 + e2);
        sprob[0] = e0 * inv; sprob[1] = e1 * inv; sprob[2] = e2 * inv;
    }
    __syncthreads();

    const float p_push = sprob[0], p_pop = sprob[1], p_noop = sprob[2];
    const size_t base = ((size_t)bIdx * CN + nIdx) * CS * CE;
    const float* si = st_in + base;
    float* so = st_out + base;

    for (int s = 0; s < CS; ++s) {
        float v;
        if (s == CS - 1) {
            v = empty_e[tid];
        } else if (s == 0) {
            v = p_push * spush[tid];
        } else {
            float dn = si[(size_t)(s - 1) * CE + tid];
            float up = si[(size_t)(s + 1) * CE + tid];
            float cu = si[(size_t)s * CE + tid];
            v = p_push * dn + p_pop * up + p_noop * cu;
        }
        so[(size_t)s * CE + tid] = v;
    }
}

// ---- finalize: copy final stacks into d_out --------------------------------
__global__ void srnn_finalize_kernel(const float* __restrict__ st_fin,
                                     float* __restrict__ out_st) {
    const int total = CB * CN * CS * CE;
    for (int i = blockIdx.x * blockDim.x + threadIdx.x; i < total;
         i += gridDim.x * blockDim.x)
        out_st[i] = st_fin[i];
}

// ---------------------------------------------------------------------------
extern "C" void kernel_launch(void* const* d_in, const int* in_sizes, int n_in,
                              void* d_out, int out_size, void* d_ws, size_t ws_size,
                              hipStream_t stream) {
    (void)in_sizes; (void)n_in; (void)out_size; (void)ws_size;

    const int*   inputs = (const int*)d_in[0];
    const float* hidden = (const float*)d_in[1];
    const float* stacks = (const float*)d_in[2];
    const float* emb    = (const float*)d_in[3];
    const float* W_ih   = (const float*)d_in[4];
    const float* W_hh   = (const float*)d_in[5];
    const float* b_ih   = (const float*)d_in[6];
    const float* b_hh   = (const float*)d_in[7];
    const float* W_act  = (const float*)d_in[8];
    const float* b_act  = (const float*)d_in[9];
    const float* W_stk  = (const float*)d_in[10];
    const float* b_stk  = (const float*)d_in[11];
    const float* emptye = (const float*)d_in[12];
    // d_in[13], d_in[14] = W_up / W_down: shift semantics hardcoded.

    float* out = (float*)d_out;            // [T*B*H] outputs | [B*H] h_final | [B*N*S*E] st_final
    char*  ws  = (char*)d_ws;

    size_t off = 0;
    __bf16* Wih_t = (__bf16*)(ws + off); off += (size_t)CH * C3H * sizeof(__bf16);
    __bf16* Whh_t = (__bf16*)(ws + off); off += (size_t)CH * C3H * sizeof(__bf16);
    __bf16* GI    = (__bf16*)(ws + off); off += (size_t)CT * CB * C3H * sizeof(__bf16);
    float*  st0   = (float*)(ws + off);  off += (size_t)CB * CN * CS * CE * sizeof(float);
    float*  st1   = (float*)(ws + off);

    float* stbuf[2] = {st0, st1};
    float* out_hfinal = out + (size_t)CT * CB * CH;
    float* out_stfinal = out_hfinal + (size_t)CB * CH;

    // 1) weight transpose/convert + stack init
    srnn_prep_kernel<<<2048, 256, 0, stream>>>(W_ih, W_hh, stacks, Wih_t, Whh_t, st0);

    // 2) big parallel input GEMM: 32768 x 1536 x 512 (bf16 WMMA, f32 acc)
    srnn_gi_gemm_kernel<<<6144, 256, 0, stream>>>(inputs, emb, Wih_t, b_ih, GI);

    // 3) sequential recurrence: stream order enforces the dependency chain
    for (int t = 0; t < CT; ++t) {
        const float* h_prev = (t == 0) ? hidden : (out + (size_t)(t - 1) * CB * CH);
        srnn_step_stack_kernel<<<CB * CN, 64, 0, stream>>>(
            h_prev, W_act, b_act, W_stk, b_stk, emptye,
            stbuf[t & 1], stbuf[(t & 1) ^ 1]);
        srnn_step_h_kernel<<<16, 256, 0, stream>>>(
            h_prev, GI + (size_t)t * CB * C3H, Whh_t, b_hh,
            out + (size_t)t * CB * CH,
            (t == CT - 1) ? out_hfinal : nullptr);
    }

    // 4) final stacks (T even -> final buffer parity 0)
    srnn_finalize_kernel<<<1024, 256, 0, stream>>>(stbuf[CT & 1], out_stfinal);
}
